// StochPhaseEvolution_24945170055792
// MI455X (gfx1250) — matrix operand order repaired and checked by reading, not compiled, not run on hardware
//
#include <hip/hip_runtime.h>

typedef __attribute__((ext_vector_type(16))) _Float16 v16h;
typedef __attribute__((ext_vector_type(8)))  float    v8f;

#define Hd 512
#define Wd 512
#define Bd 8
#define NPIX (Bd * Hd * Wd) /* 2097152 */

__device__ __forceinline__ float dev_tanh(float x) {
#if __has_builtin(__builtin_amdgcn_tanhf)
  return __builtin_amdgcn_tanhf(x);
#elif __has_builtin(__builtin_amdgcn_tanh_f32)
  return __builtin_amdgcn_tanh_f32(x);
#else
  float r;
  asm("v_tanh_f32 %0, %1" : "=v"(r) : "v"(x));
  return r;
#endif
}

// ---------------------------------------------------------------------------
// Pass A: per-pixel MLP hidden layers (tanh, TRANS-bound) with the second
// layer (K=32 dot products) done by v_wmma_f32_16x16x32_f16.
// One wave handles 16 pixels, laid out exactly as the WMMA A-matrix:
//   lane 0-15 : pixel M = lane, hidden K in {0..7, 16..23}
//   lane16-31 : pixel M = lane-16, hidden K in {8..15, 24..31}
// Writes fields: G0[p] = g0 MLP(c0[p]),
//                H0[p] = hrrp(c0[h,w], c0[h-1,w])  (edge up),
//                H1[p] = hrrp(c0[h,w], c0[h,w-1])  (edge left).
// ---------------------------------------------------------------------------
__global__ void __launch_bounds__(256) passA_kernel(
    const float* __restrict__ c,
    const float* __restrict__ g0_w1, const float* __restrict__ g0_b1,
    const float* __restrict__ g0_w2, const float* __restrict__ g0_b2,
    const float* __restrict__ h_w1,  const float* __restrict__ h_b1,
    const float* __restrict__ h_w2,  const float* __restrict__ h_b2,
    float* __restrict__ G0, float* __restrict__ H0f, float* __restrict__ H1f)
{
  const int lane = threadIdx.x & 31;
  const int wave = threadIdx.x >> 5;
  const int half = lane >> 4;   // which 16-lane half
  const int m    = lane & 15;   // pixel row M in the 16x16 tile
  const int tile = blockIdx.x * 8 + wave;  // 16 pixels per wave
  const int tileBase = tile * 16;

  // --- per-lane hidden-unit weight preload (A-layout K mapping) ------------
  // K for A-matrix VGPR v: (v<4 ? 2v : 8+2v) + 8*half, pair (K, K+1)
  float wP[16], wQ[16], wB[16], gW[16], gB[16];
#pragma unroll
  for (int v = 0; v < 8; ++v) {
    const int k0 = ((v < 4) ? 2 * v : 8 + 2 * v) + half * 8;
#pragma unroll
    for (int s = 0; s < 2; ++s) {
      const int j  = k0 + s;
      const float w0 = h_w1[j];        // h_w1[0][j]
      const float w1 = h_w1[32 + j];   // h_w1[1][j]
      wP[2 * v + s] = 0.5f * (w0 + w1);   // symmetric part
      wQ[2 * v + s] = 0.5f * (w0 - w1);   // antisymmetric part
      wB[2 * v + s] = h_b1[j];
      gW[2 * v + s] = g0_w1[j];
      gB[2 * v + s] = g0_b1[j];
    }
  }

  // --- B matrices: w2 replicated across all 16 columns ---------------------
  // 16-bit B (32x16): same K mapping as A: VGPR v holds pair
  // (K, K+1), K = (v<4 ? 2v : 8+2v) + 8*half.
  v16h Bh, Bg;
#pragma unroll
  for (int v = 0; v < 8; ++v) {
    const int kB = ((v < 4) ? 2 * v : 8 + 2 * v) + half * 8;
    Bh[2 * v]     = (_Float16)h_w2[kB];
    Bh[2 * v + 1] = (_Float16)h_w2[kB + 1];
    Bg[2 * v]     = (_Float16)g0_w2[kB];
    Bg[2 * v + 1] = (_Float16)g0_w2[kB + 1];
  }

  // --- load pixel + neighbors ----------------------------------------------
  const int pix = tileBase + m;
  const int w   = pix & (Wd - 1);
  const int h   = (pix >> 9) & (Hd - 1);
  const int b   = pix >> 18;
  const int hm  = (h == 0) ? (Hd - 1) : h - 1;
  const int wm  = (w == 0) ? (Wd - 1) : w - 1;
  const long cb = (long)b * (2 * Hd * Wd);   // c[b,0,...]
  const float a  = c[cb + (long)h * Wd + w];
  const float up = c[cb + (long)hm * Wd + w];
  const float lf = c[cb + (long)h * Wd + wm];

  // symmetrized hrrp pre-activations: tanh(u+v) + tanh(u-v)
  const float s0 = a + up, d0 = a - up;  // edge up
  const float s1 = a + lf, d1 = a - lf;  // edge left

  v16h A0, A1, Ag;
#pragma unroll
  for (int t = 0; t < 16; ++t) {
    const float u0 = fmaf(wP[t], s0, wB[t]);
    const float v0 = wQ[t] * d0;
    A0[t] = (_Float16)(dev_tanh(u0 + v0) + dev_tanh(u0 - v0));

    const float u1 = fmaf(wP[t], s1, wB[t]);
    const float v1 = wQ[t] * d1;
    A1[t] = (_Float16)(dev_tanh(u1 + v1) + dev_tanh(u1 - v1));

    Ag[t] = (_Float16)dev_tanh(fmaf(gW[t], a, gB[t]));
  }

  const v8f cz = {0.f, 0.f, 0.f, 0.f, 0.f, 0.f, 0.f, 0.f};
  // D = A x B (+0):   D[m][*] = sum_k A[m][k] * w2[k]  (all columns equal)
  v8f D0 = __builtin_amdgcn_wmma_f32_16x16x32_f16(false, A0, false, Bh,
                                                  (short)0, cz, false, false);
  v8f D1 = __builtin_amdgcn_wmma_f32_16x16x32_f16(false, A1, false, Bh,
                                                  (short)0, cz, false, false);
  v8f Dg = __builtin_amdgcn_wmma_f32_16x16x32_f16(false, Ag, false, Bg,
                                                  (short)0, cz, false, false);

  const float hb2 = h_b2[0];
  const float gb2 = g0_b2[0];

  // D layout: VGPR r, lanes 0-15 -> pixel M=r ; lanes 16-31 -> pixel M=r+8.
  // Lane 0 writes pixels 0..7, lane 16 writes pixels 8..15.
  if (m == 0) {
#pragma unroll
    for (int r = 0; r < 8; ++r) {
      const int idx = tileBase + r + 8 * half;
      G0[idx]  = Dg[r] + gb2;
      H0f[idx] = 0.5f * D0[r] + hb2;   // hrrp = 0.5*(mlp+mlp_rev); b2 outside
      H1f[idx] = 0.5f * D1[r] + hb2;
    }
  }
}

// ---------------------------------------------------------------------------
// Pass B: err = c1 - c0 - lap(G0);  OII[p] = H0[p]+H1[p]+H0[p+e0]+H1[p+e1]
// ---------------------------------------------------------------------------
__global__ void __launch_bounds__(256) passB_kernel(
    const float* __restrict__ c,
    const float* __restrict__ G0, const float* __restrict__ H0f,
    const float* __restrict__ H1f,
    float* __restrict__ ERR, float* __restrict__ OII)
{
  const int pix = blockIdx.x * blockDim.x + threadIdx.x;
  if (pix >= NPIX) return;
  const int w = pix & (Wd - 1);
  const int h = (pix >> 9) & (Hd - 1);
  const int b = pix >> 18;
  const int hm = (h == 0) ? (Hd - 1) : h - 1;
  const int hp = (h == Hd - 1) ? 0 : h + 1;
  const int wm = (w == 0) ? (Wd - 1) : w - 1;
  const int wp = (w == Wd - 1) ? 0 : w + 1;
  const int base = b << 18;  // b*H*W

  const float gc  = G0[pix];
  const float lap = G0[base + hm * Wd + w] + G0[base + hp * Wd + w] +
                    G0[base + h * Wd + wm] + G0[base + h * Wd + wp] - 4.f * gc;
  const long cb = (long)b * (2 * Hd * Wd);
  const float c0v = c[cb + (long)h * Wd + w];
  const float c1v = c[cb + (long)(Hd * Wd) + (long)h * Wd + w];
  ERR[pix] = c1v - (c0v + lap);
  // omega_ii = sum of hrrp over the 4 incident edges (sign handled in pass C)
  OII[pix] = H0f[pix] + H1f[pix] + H0f[base + hp * Wd + w] +
             H1f[base + h * Wd + wp];
}

// ---------------------------------------------------------------------------
// Pass C: per-(pixel, direction) 2x2 determinant loss; fixed-order block sum.
// ---------------------------------------------------------------------------
__global__ void __launch_bounds__(256) passC_kernel(
    const float* __restrict__ H0f, const float* __restrict__ H1f,
    const float* __restrict__ ERR, const float* __restrict__ OII,
    float* __restrict__ partial)
{
  __shared__ float sdata[256];
  const int pix = blockIdx.x * blockDim.x + threadIdx.x;
  float acc = 0.f;
  if (pix < NPIX) {
    const int w = pix & (Wd - 1);
    const int h = (pix >> 9) & (Hd - 1);
    const int b = pix >> 18;
    const int hm = (h == 0) ? (Hd - 1) : h - 1;
    const int wm = (w == 0) ? (Wd - 1) : w - 1;
    const int base = b << 18;

    const float o0  = OII[pix];
    const float e   = ERR[pix];
    const float s0v = e * e;
    {  // direction k = 0 (H axis): neighbor p - e0
      const int   q   = base + hm * Wd + w;
      const float o1  = -H0f[pix];
      const float o3  = OII[q];
      const float en  = ERR[q];
      const float det = fmaf(o0, o3, -o1 * o1);
      const float num = s0v * o3 + en * en * o0 - 2.f * (e * en) * o1;
      acc += det + num / det;
    }
    {  // direction k = 1 (W axis): neighbor p - e1
      const int   q   = base + h * Wd + wm;
      const float o1  = -H1f[pix];
      const float o3  = OII[q];
      const float en  = ERR[q];
      const float det = fmaf(o0, o3, -o1 * o1);
      const float num = s0v * o3 + en * en * o0 - 2.f * (e * en) * o1;
      acc += det + num / det;
    }
  }
  sdata[threadIdx.x] = acc;
  __syncthreads();
  for (int s = 128; s > 0; s >>= 1) {
    if (threadIdx.x < s) sdata[threadIdx.x] += sdata[threadIdx.x + s];
    __syncthreads();
  }
  if (threadIdx.x == 0) partial[blockIdx.x] = sdata[0];
}

__global__ void __launch_bounds__(256) final_kernel(
    const float* __restrict__ partial, int n, float* __restrict__ out)
{
  __shared__ float sdata[256];
  float acc = 0.f;
  for (int i = threadIdx.x; i < n; i += 256) acc += partial[i];
  sdata[threadIdx.x] = acc;
  __syncthreads();
  for (int s = 128; s > 0; s >>= 1) {
    if (threadIdx.x < s) sdata[threadIdx.x] += sdata[threadIdx.x + s];
    __syncthreads();
  }
  if (threadIdx.x == 0)
    out[0] = 0.5f * sdata[0] / (float)((long)NPIX * 2);  // 0.5 * mean
}

extern "C" void kernel_launch(void* const* d_in, const int* in_sizes, int n_in,
                              void* d_out, int out_size, void* d_ws,
                              size_t ws_size, hipStream_t stream) {
  (void)in_sizes; (void)n_in; (void)out_size; (void)ws_size;
  const float* c     = (const float*)d_in[0];
  const float* g0_w1 = (const float*)d_in[1];
  const float* g0_b1 = (const float*)d_in[2];
  const float* g0_w2 = (const float*)d_in[3];
  const float* g0_b2 = (const float*)d_in[4];
  const float* h_w1  = (const float*)d_in[5];
  const float* h_b1  = (const float*)d_in[6];
  const float* h_w2  = (const float*)d_in[7];
  const float* h_b2  = (const float*)d_in[8];

  float* ws  = (float*)d_ws;
  float* G0  = ws;
  float* H0f = ws + (size_t)NPIX;
  float* H1f = ws + (size_t)2 * NPIX;
  float* ERR = ws + (size_t)3 * NPIX;
  float* OII = ws + (size_t)4 * NPIX;
  float* prt = ws + (size_t)5 * NPIX;

  const int blocksA = NPIX / 16 / 8;  // one 16-pixel tile per wave, 8 waves/block
  passA_kernel<<<blocksA, 256, 0, stream>>>(c, g0_w1, g0_b1, g0_w2, g0_b2,
                                            h_w1, h_b1, h_w2, h_b2,
                                            G0, H0f, H1f);
  const int blocksB = NPIX / 256;     // 8192
  passB_kernel<<<blocksB, 256, 0, stream>>>(c, G0, H0f, H1f, ERR, OII);
  passC_kernel<<<blocksB, 256, 0, stream>>>(H0f, H1f, ERR, OII, prt);
  final_kernel<<<1, 256, 0, stream>>>(prt, blocksB, (float*)d_out);
}
